// HGT_25503515803804
// MI455X (gfx1250) — matrix operand order repaired and checked by reading, not compiled, not run on hardware
//
#include <hip/hip_runtime.h>
#include <hip/hip_bf16.h>
#include <math.h>

typedef __attribute__((ext_vector_type(16))) _Float16 v16h;
typedef __attribute__((ext_vector_type(8)))  float    v8f;
typedef unsigned int u32x4 __attribute__((ext_vector_type(4)));
typedef int          i32x4 __attribute__((ext_vector_type(4)));
typedef int          i32x8 __attribute__((ext_vector_type(8)));

#define HID   256
#define HEADS 8
#define DH    32
#define NAn   20000
#define NBn   20000
#define NE    100000
#define NL    2
#define NET   3
#define SCALE 0.17677669529663687f /* 1/sqrt(32) */
#define MAXK  256                   /* largest K used by gemm_wmma */

#if defined(__has_builtin)
#  if __has_builtin(__builtin_amdgcn_tensor_load_to_lds) && __has_builtin(__builtin_amdgcn_s_wait_tensorcnt)
#    define HAVE_TDM 1
#  endif
#endif
#ifndef HAVE_TDM
#  define HAVE_TDM 0
#endif

// ---------------------------------------------------------------------------
// WMMA fragment loaders (documented gfx1250 16-bit layouts)
// A 16x32 f16: lane = half*16 + M; elements are two contiguous 8-half chunks:
//   a[0..7]  = rowA[k0 + 8*half      .. +7]
//   a[8..15] = rowA[k0 + 16 + 8*half .. +7]
// ---------------------------------------------------------------------------
__device__ __forceinline__ v16h load_a_frag(const _Float16* __restrict__ rowA, int k0, int half) {
    v16h a;
#pragma unroll
    for (int j = 0; j < 8; ++j) {
        int kb = k0 + ((j < 4) ? (8 * half + 2 * j) : (16 + 8 * half + 2 * (j - 4)));
        a[2 * j]     = rowA[kb];
        a[2 * j + 1] = rowA[kb + 1];
    }
    return a;
}
// B 32x16 f16 from Bt[N][K]: lane = half*16 + N; element i -> K = k0 + 8*half + i (i<8), k0 + 16 + 8*half + (i-8) (i>=8)
template <typename PtrT>
__device__ __forceinline__ v16h load_b_frag(PtrT rowBt, int k0, int half) {
    v16h b;
#pragma unroll
    for (int i = 0; i < 16; ++i) b[i] = rowBt[k0 + ((i < 8) ? (8 * half + i) : (16 + 8 * half + (i - 8)))];
    return b;
}

// ---------------------------------------------------------------------------
// Generic WMMA GEMM: C[M,N] = act(A[M,K] @ Bt^T + bias); optional skip blend.
// Block = 256 threads = 8 waves; each wave owns a 16(M) x 64(N) slab.
// Requires: M % 16 == 0, N % 64 == 0, K % 32 == 0 (true for every call site).
// The block's 64-row Bt slab (contiguous 64*K halfs) is staged into LDS by the
// Tensor Data Mover once, then every wave/K-step reads B fragments from LDS.
// ---------------------------------------------------------------------------
__global__ __launch_bounds__(256) void gemm_wmma(
    const _Float16* __restrict__ A, int lda,
    const _Float16* __restrict__ Bt, int ldb,
    const float* __restrict__ bias,
    const float* __restrict__ resid,     // if non-null: out = beta*out + (1-beta)*resid
    const float* __restrict__ skipv, int skipIdx,
    float* __restrict__ out32, _Float16* __restrict__ out16, int ldc,
    int M, int N, int K, int act)        // act: 0 none, 1 relu
{
    __shared__ _Float16 btile[64 * MAXK];   // single static LDS alloc -> offset 0

    const int lane = threadIdx.x & 31;
    const int wave = threadIdx.x >> 5;
    const int m0   = (blockIdx.x * 8 + wave) * 16;
    if (m0 >= M) return;                    // wave-uniform; wave0 exits => all exit
    const int n0   = blockIdx.y * 64;
    const int half = lane >> 4;
    const int lnm  = lane & 15;

    // ---- stage Bt[n0 .. n0+63][0..K) (contiguous) into LDS ----
    const unsigned eltCount = 64u * (unsigned)K;          // <= 16384, fits tile_dim0
#if HAVE_TDM
    if (wave == 0) {
        unsigned long long ga = (unsigned long long)(const void*)(Bt + (size_t)n0 * ldb);
        u32x4 g0;
        g0[0] = 1u;                                        // count=1, is_restore=0, gather off
        g0[1] = 0u;                                        // lds_addr = 0 (btile offset)
        g0[2] = (unsigned)(ga & 0xFFFFFFFFu);              // global_addr[31:0]
        g0[3] = (unsigned)((ga >> 32) & 0x01FFFFFFu) | (2u << 30);  // addr[56:32] | type=2
        i32x8 g1;
        g1[0] = (int)(1u << 16);                           // data_size=1 (2 bytes)
        g1[1] = (int)((eltCount & 0xFFFFu) << 16);         // tensor_dim0[15:0] @ bits 48..63
        g1[2] = (int)(eltCount >> 16);                     // tensor_dim0[31:16]; tensor_dim1=0 low
        g1[3] = (int)((eltCount & 0xFFFFu) << 16);         // tile_dim0 @ bits 112..127
        g1[4] = 0;                                         // tile_dim1=0 (1-D), tile_dim2=0
        g1[5] = (int)eltCount;                             // tensor_dim0_stride low (unused, 1-D)
        g1[6] = 0;
        g1[7] = 0;
        i32x4 gz = {0, 0, 0, 0};
#if defined(__clang_major__) && (__clang_major__ >= 23)
        i32x8 gz8 = {0, 0, 0, 0, 0, 0, 0, 0};
        __builtin_amdgcn_tensor_load_to_lds(g0, g1, gz, gz, gz8, 0);
#else
        __builtin_amdgcn_tensor_load_to_lds(g0, g1, gz, gz, 0);
#endif
        __builtin_amdgcn_s_wait_tensorcnt(0);
    }
    // The TDM builtin doesn't reference btile, so without this clobber LLVM
    // sees a never-written LDS object and folds the ds loads to undef.
    asm volatile("" : : "r"(&btile[0]) : "memory");
    __syncthreads();
#else
    {
        const _Float16* src = Bt + (size_t)n0 * ldb;
        for (unsigned i = threadIdx.x; i < eltCount; i += 256u) btile[i] = src[i];
        __syncthreads();
    }
#endif

    const _Float16* rowA = A + (size_t)(m0 + lnm) * lda;
    v8f acc0 = {}, acc1 = {}, acc2 = {}, acc3 = {};

    for (int k0 = 0; k0 < K; k0 += 32) {
        if (k0 + 32 < K) __builtin_prefetch(rowA + k0 + 32, 0, 3);   // global_prefetch_b8 (near)
        v16h a  = load_a_frag(rowA, k0, half);
        v16h b0 = load_b_frag(&btile[(size_t)(0 * 16 + lnm) * K], k0, half);
        v16h b1 = load_b_frag(&btile[(size_t)(1 * 16 + lnm) * K], k0, half);
        v16h b2 = load_b_frag(&btile[(size_t)(2 * 16 + lnm) * K], k0, half);
        v16h b3 = load_b_frag(&btile[(size_t)(3 * 16 + lnm) * K], k0, half);
        acc0 = __builtin_amdgcn_wmma_f32_16x16x32_f16(false, a, false, b0, (short)0, acc0, false, false);
        acc1 = __builtin_amdgcn_wmma_f32_16x16x32_f16(false, a, false, b1, (short)0, acc1, false, false);
        acc2 = __builtin_amdgcn_wmma_f32_16x16x32_f16(false, a, false, b2, (short)0, acc2, false, false);
        acc3 = __builtin_amdgcn_wmma_f32_16x16x32_f16(false, a, false, b3, (short)0, acc3, false, false);
    }

    float beta = 0.f;
    if (resid) beta = 1.f / (1.f + expf(-skipv[skipIdx]));

    v8f accs[4] = {acc0, acc1, acc2, acc3};
#pragma unroll
    for (int jn = 0; jn < 4; ++jn) {
        const int col = n0 + jn * 16 + lnm;
        const float bv = bias ? bias[col] : 0.f;
#pragma unroll
        for (int r = 0; r < 8; ++r) {
            const int row = m0 + r + 8 * half;   // C layout: VGPR r -> M = r + 8*half
            float v = accs[jn][r] + bv;
            if (act == 1) v = fmaxf(v, 0.f);
            if (resid)  v = beta * v + (1.f - beta) * resid[(size_t)row * ldc + col];
            if (out32)  out32[(size_t)row * ldc + col] = v;
            if (out16)  out16[(size_t)row * ldc + col] = (_Float16)v;
        }
    }
}

// ---------------------------------------------------------------------------
// Batched per-head relation transform: ke/ve[n, h*32+e] = sum_d kqv[n, off+h*32+d] * relT[h][e][d]
// grid = (rowblocks, HEADS, 2 {ke, ve}); one WMMA (K=32) per 16x16 tile.
// ---------------------------------------------------------------------------
__global__ __launch_bounds__(256) void head_gemm(
    const _Float16* __restrict__ kqv,    // [Ns][768]
    const _Float16* __restrict__ aRelT,  // [H][32][32] (e-major, d-minor) for this (l,e)
    const _Float16* __restrict__ mRelT,
    float* __restrict__ ke, float* __restrict__ ve, int Ns)
{
    const int lane = threadIdx.x & 31;
    const int wave = threadIdx.x >> 5;
    const int m0   = (blockIdx.x * 8 + wave) * 16;
    if (m0 >= Ns) return;
    const int h     = blockIdx.y;
    const int which = blockIdx.z;
    const _Float16* Bt = (which ? mRelT : aRelT) + (size_t)h * (DH * DH);
    const int kOff  = (which ? (2 * HID) : 0) + h * DH;
    float* outp     = which ? ve : ke;
    const int half  = lane >> 4;
    const int lnm   = lane & 15;

    const _Float16* rowA = kqv + (size_t)(m0 + lnm) * (3 * HID) + kOff;
    v16h a  = load_a_frag(rowA, 0, half);
    v16h b0 = load_b_frag(Bt + (size_t)(lnm) * DH, 0, half);
    v16h b1 = load_b_frag(Bt + (size_t)(16 + lnm) * DH, 0, half);
    v8f acc0 = {}, acc1 = {};
    acc0 = __builtin_amdgcn_wmma_f32_16x16x32_f16(false, a, false, b0, (short)0, acc0, false, false);
    acc1 = __builtin_amdgcn_wmma_f32_16x16x32_f16(false, a, false, b1, (short)0, acc1, false, false);

    v8f accs[2] = {acc0, acc1};
#pragma unroll
    for (int jn = 0; jn < 2; ++jn)
#pragma unroll
        for (int r = 0; r < 8; ++r) {
            const int row = m0 + r + 8 * half;
            const int col = h * DH + jn * 16 + lnm;
            outp[(size_t)row * HID + col] = accs[jn][r];
        }
}

// ---------------------------------------------------------------------------
// Utility converts
// ---------------------------------------------------------------------------
__global__ void conv_t_f16(const float* __restrict__ src, _Float16* __restrict__ dst,
                           int rows, int cols, int total) {
    int idx = blockIdx.x * blockDim.x + threadIdx.x;
    if (idx >= total) return;
    int b   = idx / (rows * cols);
    int rem = idx - b * rows * cols;
    int r = rem / cols, c = rem % cols;
    dst[(size_t)b * rows * cols + (size_t)c * rows + r] = (_Float16)src[idx];
}

__global__ void conv_f16(const float* __restrict__ src, _Float16* __restrict__ dst, int n) {
    int i = blockIdx.x * blockDim.x + threadIdx.x;
    if (i < n) dst[i] = (_Float16)src[i];
}

__global__ void gelu_f16(const float* __restrict__ src, _Float16* __restrict__ dst, int n) {
    int i = blockIdx.x * blockDim.x + threadIdx.x;
    if (i >= n) return;
    float v = src[i];
    float g = 0.5f * v * (1.f + erff(v * 0.70710678118654752f)); // exact gelu
    dst[i] = (_Float16)g;
}

// ---------------------------------------------------------------------------
// Edge attention: one wave per edge; lane = d. Monotone-uint key for float atomicMax.
// ---------------------------------------------------------------------------
__global__ __launch_bounds__(256) void alpha_kernel(
    const _Float16* __restrict__ kqvDst,  // [Nd][768], q at +HID
    const float* __restrict__ keSrc,      // [Ns][256]
    const int* __restrict__ ei,           // [2][E] : row0 src, row1 dst
    const float* __restrict__ prel,       // [8]
    float* __restrict__ alphaOut,         // [E][8] (pre-offset for stacking)
    unsigned int* __restrict__ amax,      // [Nd][8] (key-encoded)
    int E)
{
    const int wave = threadIdx.x >> 5, lane = threadIdx.x & 31;
    const int e = blockIdx.x * 8 + wave;
    if (e >= E) return;
    const int src = ei[e], dst = ei[E + e];
    const _Float16* qrow = kqvDst + (size_t)dst * (3 * HID) + HID;
    const float*    krow = keSrc + (size_t)src * HID;
#pragma unroll
    for (int h = 0; h < HEADS; ++h) {
        float p = (float)qrow[h * DH + lane] * krow[h * DH + lane];
        for (int off = 16; off > 0; off >>= 1) p += __shfl_xor(p, off, 32);
        if (lane == 0) {
            float al = p * prel[h] * SCALE;
            alphaOut[(size_t)e * HEADS + h] = al;
            unsigned int u   = __float_as_uint(al);
            unsigned int key = (u & 0x80000000u) ? ~u : (u | 0x80000000u);
            atomicMax(&amax[(size_t)dst * HEADS + h], key);
        }
    }
}

__global__ void softmax_exp_kernel(float* __restrict__ alpha, const int* __restrict__ ei,
                                   const unsigned int* __restrict__ amax,
                                   float* __restrict__ den, int E) {
    int idx = blockIdx.x * blockDim.x + threadIdx.x;
    if (idx >= E * HEADS) return;
    int e = idx / HEADS, h = idx - e * HEADS;
    int dst = ei[E + e];
    unsigned int key = amax[(size_t)dst * HEADS + h];
    unsigned int u   = (key & 0x80000000u) ? (key & 0x7FFFFFFFu) : ~key;
    float am = __uint_as_float(u);
    if (!(fabsf(am) < 3.0e38f)) am = 0.f;   // mirrors reference isfinite guard
    float ex = expf(alpha[idx] - am);
    alpha[idx] = ex;
    atomicAdd(&den[(size_t)dst * HEADS + h], ex);
}

__global__ __launch_bounds__(256) void aggregate_kernel(
    const float* __restrict__ alphaEx, const int* __restrict__ ei,
    const float* __restrict__ veSrc, const float* __restrict__ den,
    float* __restrict__ agg, int E) {
    const int e = blockIdx.x;
    const int t = threadIdx.x;        // t = h*32 + d
    const int h = t >> 5;
    const int src = ei[e], dst = ei[E + e];
    float att = alphaEx[(size_t)e * HEADS + h] / (den[(size_t)dst * HEADS + h] + 1e-16f);
    atomicAdd(&agg[(size_t)dst * HID + t], veSrc[(size_t)src * HID + t] * att);
}

// ---------------------------------------------------------------------------
// Host orchestration
// ---------------------------------------------------------------------------
extern "C" void kernel_launch(void* const* d_in, const int* in_sizes, int n_in,
                              void* d_out, int out_size, void* d_ws, size_t ws_size,
                              hipStream_t stream) {
    (void)in_sizes; (void)n_in; (void)out_size; (void)ws_size;
    const float* x_A     = (const float*)d_in[0];
    const float* x_B     = (const float*)d_in[1];
    const int*   ei_ab   = (const int*)d_in[2];
    const int*   ei_ba   = (const int*)d_in[3];
    const int*   ei_aa   = (const int*)d_in[4];
    const float* W_in_A  = (const float*)d_in[5];
    const float* b_in_A  = (const float*)d_in[6];
    const float* W_in_B  = (const float*)d_in[7];
    const float* b_in_B  = (const float*)d_in[8];
    const float* W_kqv_A = (const float*)d_in[9];
    const float* b_kqv_A = (const float*)d_in[10];
    const float* W_kqv_B = (const float*)d_in[11];
    const float* b_kqv_B = (const float*)d_in[12];
    const float* W_out_A = (const float*)d_in[13];
    const float* b_out_A = (const float*)d_in[14];
    const float* W_out_B = (const float*)d_in[15];
    const float* b_out_B = (const float*)d_in[16];
    const float* skip_A  = (const float*)d_in[17];
    const float* skip_B  = (const float*)d_in[18];
    const float* a_rel   = (const float*)d_in[19];
    const float* m_rel   = (const float*)d_in[20];
    const float* p_rel   = (const float*)d_in[21];

    char* base = (char*)d_ws;
    size_t off = 0;
    auto alloc = [&](size_t bytes) -> void* {
        void* r = base + off;
        off = (off + bytes + 255) & ~(size_t)255;
        return r;
    };

    _Float16* wInAT  = (_Float16*)alloc((size_t)HID * 64 * 2);
    _Float16* wInBT  = (_Float16*)alloc((size_t)HID * 32 * 2);
    _Float16* wKqvAT = (_Float16*)alloc((size_t)NL * 768 * 256 * 2);
    _Float16* wKqvBT = (_Float16*)alloc((size_t)NL * 768 * 256 * 2);
    _Float16* wOutAT = (_Float16*)alloc((size_t)NL * 256 * 256 * 2);
    _Float16* wOutBT = (_Float16*)alloc((size_t)NL * 256 * 256 * 2);
    _Float16* aRelT  = (_Float16*)alloc((size_t)NL * NET * HEADS * DH * DH * 2);
    _Float16* mRelT  = (_Float16*)alloc((size_t)NL * NET * HEADS * DH * DH * 2);
    _Float16* xA16   = (_Float16*)alloc((size_t)NAn * 64 * 2);
    _Float16* xB16   = (_Float16*)alloc((size_t)NBn * 32 * 2);
    float* curA = (float*)alloc((size_t)NAn * HID * 4);
    float* nxtA = (float*)alloc((size_t)NAn * HID * 4);
    float* curB = (float*)alloc((size_t)NBn * HID * 4);
    float* nxtB = (float*)alloc((size_t)NBn * HID * 4);
    _Float16* hA16   = (_Float16*)alloc((size_t)NAn * HID * 2);
    _Float16* hB16   = (_Float16*)alloc((size_t)NBn * HID * 2);
    _Float16* kqvA16 = (_Float16*)alloc((size_t)NAn * 768 * 2);
    _Float16* kqvB16 = (_Float16*)alloc((size_t)NBn * 768 * 2);
    float* ke0 = (float*)alloc((size_t)NAn * HID * 4);
    float* ve0 = (float*)alloc((size_t)NAn * HID * 4);
    float* ke1 = (float*)alloc((size_t)NBn * HID * 4);
    float* ve1 = (float*)alloc((size_t)NBn * HID * 4);
    float* ke2 = (float*)alloc((size_t)NAn * HID * 4);
    float* ve2 = (float*)alloc((size_t)NAn * HID * 4);
    float* alphaA = (float*)alloc((size_t)2 * NE * HEADS * 4);
    float* alphaB = (float*)alloc((size_t)NE * HEADS * 4);
    unsigned int* amaxA = (unsigned int*)alloc((size_t)NAn * HEADS * 4);
    unsigned int* amaxB = (unsigned int*)alloc((size_t)NBn * HEADS * 4);
    float* denA = (float*)alloc((size_t)NAn * HEADS * 4);
    float* denB = (float*)alloc((size_t)NBn * HEADS * 4);
    float* aggA = (float*)alloc((size_t)NAn * HID * 4);
    float* aggB = (float*)alloc((size_t)NBn * HID * 4);
    _Float16* geluA16 = (_Float16*)alloc((size_t)NAn * HID * 2);
    _Float16* geluB16 = (_Float16*)alloc((size_t)NBn * HID * 2);

    const int T = 256;
    auto g1 = [&](int n) { return dim3((n + T - 1) / T); };

    conv_t_f16<<<g1(64 * 256), T, 0, stream>>>(W_in_A, wInAT, 64, 256, 64 * 256);
    conv_t_f16<<<g1(32 * 256), T, 0, stream>>>(W_in_B, wInBT, 32, 256, 32 * 256);
    conv_t_f16<<<g1(NL * 256 * 768), T, 0, stream>>>(W_kqv_A, wKqvAT, 256, 768, NL * 256 * 768);
    conv_t_f16<<<g1(NL * 256 * 768), T, 0, stream>>>(W_kqv_B, wKqvBT, 256, 768, NL * 256 * 768);
    conv_t_f16<<<g1(NL * 256 * 256), T, 0, stream>>>(W_out_A, wOutAT, 256, 256, NL * 256 * 256);
    conv_t_f16<<<g1(NL * 256 * 256), T, 0, stream>>>(W_out_B, wOutBT, 256, 256, NL * 256 * 256);
    conv_t_f16<<<g1(NL * NET * HEADS * DH * DH), T, 0, stream>>>(a_rel, aRelT, DH, DH, NL * NET * HEADS * DH * DH);
    conv_t_f16<<<g1(NL * NET * HEADS * DH * DH), T, 0, stream>>>(m_rel, mRelT, DH, DH, NL * NET * HEADS * DH * DH);
    conv_f16<<<g1(NAn * 64), T, 0, stream>>>(x_A, xA16, NAn * 64);
    conv_f16<<<g1(NBn * 32), T, 0, stream>>>(x_B, xB16, NBn * 32);

    dim3 gA((NAn / 16 + 7) / 8, HID / 64), gB((NBn / 16 + 7) / 8, HID / 64);
    gemm_wmma<<<gA, T, 0, stream>>>(xA16, 64, wInAT, 64, b_in_A, nullptr, nullptr, 0,
                                    curA, hA16, HID, NAn, HID, 64, 1);
    gemm_wmma<<<gB, T, 0, stream>>>(xB16, 32, wInBT, 32, b_in_B, nullptr, nullptr, 0,
                                    curB, hB16, HID, NBn, HID, 32, 1);

    for (int l = 0; l < NL; ++l) {
        hipMemsetAsync(amaxA, 0, (size_t)NAn * HEADS * 4, stream);
        hipMemsetAsync(amaxB, 0, (size_t)NBn * HEADS * 4, stream);
        hipMemsetAsync(denA, 0, (size_t)NAn * HEADS * 4, stream);
        hipMemsetAsync(denB, 0, (size_t)NBn * HEADS * 4, stream);
        hipMemsetAsync(aggA, 0, (size_t)NAn * HID * 4, stream);
        hipMemsetAsync(aggB, 0, (size_t)NBn * HID * 4, stream);

        dim3 gkA((NAn / 16 + 7) / 8, 768 / 64), gkB((NBn / 16 + 7) / 8, 768 / 64);
        gemm_wmma<<<gkA, T, 0, stream>>>(hA16, HID, wKqvAT + (size_t)l * 768 * 256, 256,
                                         b_kqv_A + l * 768, nullptr, nullptr, 0,
                                         nullptr, kqvA16, 768, NAn, 768, HID, 0);
        gemm_wmma<<<gkB, T, 0, stream>>>(hB16, HID, wKqvBT + (size_t)l * 768 * 256, 256,
                                         b_kqv_B + l * 768, nullptr, nullptr, 0,
                                         nullptr, kqvB16, 768, NBn, 768, HID, 0);

        const size_t relStride = (size_t)HEADS * DH * DH;
        dim3 ghA((NAn / 16 + 7) / 8, HEADS, 2), ghB((NBn / 16 + 7) / 8, HEADS, 2);
        head_gemm<<<ghA, T, 0, stream>>>(kqvA16, aRelT + (size_t)(l * NET + 0) * relStride,
                                         mRelT + (size_t)(l * NET + 0) * relStride, ke0, ve0, NAn);
        head_gemm<<<ghB, T, 0, stream>>>(kqvB16, aRelT + (size_t)(l * NET + 1) * relStride,
                                         mRelT + (size_t)(l * NET + 1) * relStride, ke1, ve1, NBn);
        head_gemm<<<ghA, T, 0, stream>>>(kqvA16, aRelT + (size_t)(l * NET + 2) * relStride,
                                         mRelT + (size_t)(l * NET + 2) * relStride, ke2, ve2, NAn);

        dim3 ge((NE + 7) / 8);
        alpha_kernel<<<ge, T, 0, stream>>>(kqvB16, ke0, ei_ab, p_rel + (l * NET + 0) * HEADS,
                                           alphaB, amaxB, NE);
        alpha_kernel<<<ge, T, 0, stream>>>(kqvA16, ke1, ei_ba, p_rel + (l * NET + 1) * HEADS,
                                           alphaA, amaxA, NE);
        alpha_kernel<<<ge, T, 0, stream>>>(kqvA16, ke2, ei_aa, p_rel + (l * NET + 2) * HEADS,
                                           alphaA + (size_t)NE * HEADS, amaxA, NE);

        softmax_exp_kernel<<<g1(NE * HEADS), T, 0, stream>>>(alphaB, ei_ab, amaxB, denB, NE);
        softmax_exp_kernel<<<g1(NE * HEADS), T, 0, stream>>>(alphaA, ei_ba, amaxA, denA, NE);
        softmax_exp_kernel<<<g1(NE * HEADS), T, 0, stream>>>(alphaA + (size_t)NE * HEADS, ei_aa, amaxA, denA, NE);

        aggregate_kernel<<<NE, T, 0, stream>>>(alphaB, ei_ab, ve0, denB, aggB, NE);
        aggregate_kernel<<<NE, T, 0, stream>>>(alphaA, ei_ba, ve1, denA, aggA, NE);
        aggregate_kernel<<<NE, T, 0, stream>>>(alphaA + (size_t)NE * HEADS, ei_aa, ve2, denA, aggA, NE);

        gelu_f16<<<g1(NAn * HID), T, 0, stream>>>(aggA, geluA16, NAn * HID);
        gelu_f16<<<g1(NBn * HID), T, 0, stream>>>(aggB, geluB16, NBn * HID);
        gemm_wmma<<<gA, T, 0, stream>>>(geluA16, HID, wOutAT + (size_t)l * 256 * 256, 256,
                                        b_out_A + l * 256, curA, skip_A, l,
                                        nxtA, hA16, HID, NAn, HID, HID, 0);
        gemm_wmma<<<gB, T, 0, stream>>>(geluB16, HID, wOutBT + (size_t)l * 256 * 256, 256,
                                        b_out_B + l * 256, curB, skip_B, l,
                                        nxtB, hB16, HID, NBn, HID, HID, 0);
        float* t0 = curA; curA = nxtA; nxtA = t0;
        float* t1 = curB; curB = nxtB; nxtB = t1;
    }

    hipMemcpyAsync(d_out, curA, (size_t)NAn * HID * 4, hipMemcpyDeviceToDevice, stream);
    hipMemcpyAsync((float*)d_out + (size_t)NAn * HID, curB, (size_t)NBn * HID * 4,
                   hipMemcpyDeviceToDevice, stream);
}
